// MomentumSSM_18219251270232
// MI455X (gfx1250) — compile-verified, matrix operand order checked by reading, not run on hardware
//
#include <hip/hip_runtime.h>
#include <math.h>
#include <stdint.h>

// CDNA5 / gfx1250: wave32, WMMA 16x16x32 f16 -> f32
typedef __attribute__((ext_vector_type(16))) _Float16 v16h;
typedef __attribute__((ext_vector_type(8)))  float    v8f;

#define D_IN   512
#define D_ST   16
#define DTRK   32
#define BSZ    2
#define LSEQ   4096
#define NPROJ  64                 // DTRK + 2*D_ST
#define ROWS   (BSZ*LSEQ)         // 8192
#define CHUNK  64
#define NCH    (LSEQ/CHUNK)       // 64

// K index held by VGPR-half v2 (0..7) for the 16-bit A/B 16x32 WMMA layout.
// lanes 0-15 (lh=0): V0..3 -> K 0..7, V4..7 -> K 16..23 ; lanes 16-31 (+8)
__device__ __forceinline__ int wmma_kb(int v2, int lh) {
  return (v2 < 4 ? 2 * v2 : 16 + 2 * (v2 - 4)) + lh * 8;
}

__device__ __forceinline__ float sigmoidf_dev(float z) {
  return 1.0f / (1.0f + __expf(-z));
}

// Async copy: LDS[laddr] = *(base + goff), tracked by ASYNCcnt (CDNA5 path,
// bypasses the VGPR round-trip of load+ds_store).
__device__ __forceinline__ void async_g2l_b32(uint32_t laddr, uint32_t goff,
                                              const void* base) {
  asm volatile("global_load_async_to_lds_b32 %0, %1, %2"
               :: "v"(laddr), "v"(goff), "s"(base) : "memory");
}
__device__ __forceinline__ void wait_asynccnt0() {
  asm volatile("s_wait_asynccnt 0" ::: "memory");
}
__device__ __forceinline__ uint32_t lds_addr_of(const void* p) {
  return (uint32_t)(uintptr_t)p;   // LDS aperture: low 32 bits = LDS offset
}

// ---------------------------------------------------------------- prep ----
__global__ __launch_bounds__(256)
void prep_kernel(const float* __restrict__ Wx, const float* __restrict__ Wdt,
                 _Float16* __restrict__ wx16, _Float16* __restrict__ wdt16) {
  int i = blockIdx.x * blockDim.x + threadIdx.x;
  if (i < NPROJ * D_IN) wx16[i] = (_Float16)Wx[i];
  int j = i - NPROJ * D_IN;
  if (j >= 0 && j < D_IN * DTRK) wdt16[j] = (_Float16)Wdt[j];
}

// ------------------------------------------------- GEMM1: proj = x @ Wxᵀ --
// M=8192, N=64, K=512. One wave -> 16 rows x 64 cols, K-loop step 32.
__global__ __launch_bounds__(256)
void gemm_proj(const float* __restrict__ x, const _Float16* __restrict__ wx16,
               float* __restrict__ proj) {
  const int wave = threadIdx.x >> 5, lane = threadIdx.x & 31;
  const int lm = lane & 15, lh = lane >> 4;
  const int m0 = (blockIdx.x * 8 + wave) * 16;
  const int m  = m0 + lm;
  v8f acc[4];
  #pragma unroll
  for (int nt = 0; nt < 4; ++nt) acc[nt] = (v8f){};
  for (int kk = 0; kk < D_IN; kk += 32) {
    v16h a;
    #pragma unroll
    for (int v2 = 0; v2 < 8; ++v2) {
      const int kb = kk + wmma_kb(v2, lh);
      a[2 * v2]     = (_Float16)x[(size_t)m * D_IN + kb];
      a[2 * v2 + 1] = (_Float16)x[(size_t)m * D_IN + kb + 1];
    }
    #pragma unroll
    for (int nt = 0; nt < 4; ++nt) {
      v16h bf;
      const int n = nt * 16 + lm;
      #pragma unroll
      for (int v2 = 0; v2 < 8; ++v2) {
        const int kb = kk + wmma_kb(v2, lh);
        bf[2 * v2]     = wx16[(size_t)n * D_IN + kb];
        bf[2 * v2 + 1] = wx16[(size_t)n * D_IN + kb + 1];
      }
      acc[nt] = __builtin_amdgcn_wmma_f32_16x16x32_f16(
          false, a, false, bf, (short)0, acc[nt], false, false);
    }
  }
  // C/D layout: VGPR vv -> row m0+vv (+8 for lanes 16-31), col = lane%16
  #pragma unroll
  for (int nt = 0; nt < 4; ++nt)
    #pragma unroll
    for (int vv = 0; vv < 8; ++vv) {
      const int row = m0 + vv + 8 * lh;
      const int col = nt * 16 + lm;
      proj[(size_t)row * NPROJ + col] = acc[nt][vv];
    }
}

// ------------------------------- GEMM2: dt = softplus(proj[:,:32]@Wdtᵀ+b) --
// M=8192, N=512, K=32 (single WMMA per tile pair).
__global__ __launch_bounds__(256)
void gemm_dt(const float* __restrict__ proj, const _Float16* __restrict__ wdt16,
             const float* __restrict__ b_dt, float* __restrict__ dtbuf) {
  const int wave = threadIdx.x >> 5, lane = threadIdx.x & 31;
  const int lm = lane & 15, lh = lane >> 4;
  const int m0 = blockIdx.x * 16;
  const int n0 = wave * 64;
  const int m  = m0 + lm;
  v16h a;
  #pragma unroll
  for (int v2 = 0; v2 < 8; ++v2) {
    const int kb = wmma_kb(v2, lh);
    a[2 * v2]     = (_Float16)proj[(size_t)m * NPROJ + kb];
    a[2 * v2 + 1] = (_Float16)proj[(size_t)m * NPROJ + kb + 1];
  }
  #pragma unroll
  for (int nt = 0; nt < 4; ++nt) {
    v16h bf;
    const int n = n0 + nt * 16 + lm;
    #pragma unroll
    for (int v2 = 0; v2 < 8; ++v2) {
      const int kb = wmma_kb(v2, lh);
      bf[2 * v2]     = wdt16[(size_t)n * DTRK + kb];
      bf[2 * v2 + 1] = wdt16[(size_t)n * DTRK + kb + 1];
    }
    v8f acc = (v8f){};
    acc = __builtin_amdgcn_wmma_f32_16x16x32_f16(
        false, a, false, bf, (short)0, acc, false, false);
    #pragma unroll
    for (int vv = 0; vv < 8; ++vv) {
      const int row = m0 + vv + 8 * lh;
      const int col = n0 + nt * 16 + lm;
      const float z = acc[vv] + b_dt[col];
      const float sp = (z > 20.0f) ? z : log1pf(__expf(z));
      dtbuf[(size_t)row * D_IN + col] = sp;
    }
  }
}

// ----------------------------------------- phase A: per-chunk local scan --
// One lane owns one (b, d, chunk); 16 states in registers.
// Tracks local state (h,v from 0) and chunk transition P11, P12 (P22 = beta^t).
__global__ __launch_bounds__(512)
void scan_phaseA(const float* __restrict__ x, const float* __restrict__ proj,
                 const float* __restrict__ dtbuf, const float* __restrict__ A_log,
                 const float* __restrict__ p_alpha, const float* __restrict__ p_blogit,
                 float* __restrict__ summ) {
  __shared__ float sBp[CHUNK * D_ST];
  const int b = blockIdx.x / NCH;
  const int c = blockIdx.x % NCH;
  const int d = threadIdx.x;
  const int rbase = b * LSEQ + c * CHUNK;
  // Issue async Bp staging (global -> LDS direct), overlap with setup below.
  {
    const uint32_t lb = lds_addr_of(&sBp[0]);
    for (int i = threadIdx.x; i < CHUNK * D_ST; i += blockDim.x) {
      const uint32_t goff =
          (uint32_t)(((rbase + (i >> 4)) * NPROJ + DTRK + (i & 15)) * 4);
      async_g2l_b32(lb + (uint32_t)i * 4u, goff, proj);
    }
  }
  const float alpha = p_alpha[0];
  const float beta  = sigmoidf_dev(p_blogit[0]);
  float h[D_ST], v[D_ST], p11[D_ST], p12[D_ST], Ac[D_ST];
  #pragma unroll
  for (int n = 0; n < D_ST; ++n) {
    h[n] = 0.f; v[n] = 0.f; p11[n] = 1.f; p12[n] = 0.f;
    Ac[n] = -__expf(A_log[d * D_ST + n]);
  }
  wait_asynccnt0();
  __syncthreads();
  float P22 = 1.f;
  for (int t = 0; t < CHUNK; ++t) {
    const int r = rbase + t;
    const float dtv = dtbuf[(size_t)r * D_IN + d];
    const float xv  = x[(size_t)r * D_IN + d];
    __builtin_prefetch(&dtbuf[(size_t)(r + 16) * D_IN + d], 0, 0);
    __builtin_prefetch(&x[(size_t)(r + 16) * D_IN + d], 0, 0);
    const float s = alpha * dtv * xv;
    #pragma unroll
    for (int n = 0; n < D_ST; ++n) {
      const float at = __expf(dtv * Ac[n]);
      v[n]   = beta * v[n] + s * sBp[t * D_ST + n];
      h[n]   = at * h[n] + v[n];
      p12[n] = at * p12[n] + beta * P22;
      p11[n] *= at;
    }
    P22 *= beta;
  }
  const size_t base = ((size_t)(b * NCH + c) * D_IN + d) * 64;
  #pragma unroll
  for (int n = 0; n < D_ST; ++n) {
    summ[base + n]      = h[n];
    summ[base + 16 + n] = v[n];
    summ[base + 32 + n] = p11[n];
    summ[base + 48 + n] = p12[n];
  }
}

// -------------------------------- phase B: sequential carry over chunks --
__global__ __launch_bounds__(256)
void scan_phaseB(const float* __restrict__ summ, const float* __restrict__ p_blogit,
                 float* __restrict__ carry) {
  const int tid = blockIdx.x * blockDim.x + threadIdx.x; // 0..16383
  const int n = tid & 15;
  const int d = (tid >> 4) & (D_IN - 1);
  const int b = tid >> 13;
  const float beta = sigmoidf_dev(p_blogit[0]);
  float betaC = beta;                 // beta^64 by squaring
  #pragma unroll
  for (int i = 0; i < 6; ++i) betaC *= betaC;
  float hin = 0.f, vin = 0.f;
  for (int c = 0; c < NCH; ++c) {
    const size_t sb = ((size_t)(b * NCH + c) * D_IN + d) * 64 + n;
    const size_t cb = ((size_t)(b * NCH + c) * D_IN + d) * 32 + n;
    carry[cb]      = hin;            // state at chunk start
    carry[cb + 16] = vin;
    const float hl  = summ[sb];
    const float vl  = summ[sb + 16];
    const float q11 = summ[sb + 32];
    const float q12 = summ[sb + 48];
    const float hn = q11 * hin + q12 * vin + hl;
    const float vn = betaC * vin + vl;
    hin = hn; vin = vn;
  }
}

// ------------------------- phase C: re-scan with carry, emit y + D*x ----
__global__ __launch_bounds__(512)
void scan_phaseC(const float* __restrict__ x, const float* __restrict__ proj,
                 const float* __restrict__ dtbuf, const float* __restrict__ A_log,
                 const float* __restrict__ D_prm, const float* __restrict__ p_alpha,
                 const float* __restrict__ p_blogit, const float* __restrict__ carry,
                 float* __restrict__ out) {
  __shared__ float sBp[CHUNK * D_ST];
  __shared__ float sCp[CHUNK * D_ST];
  const int b = blockIdx.x / NCH;
  const int c = blockIdx.x % NCH;
  const int d = threadIdx.x;
  const int rbase = b * LSEQ + c * CHUNK;
  // Async staging of Bp and Cp chunks into LDS; overlap with carry/Ac setup.
  {
    const uint32_t lB = lds_addr_of(&sBp[0]);
    const uint32_t lC = lds_addr_of(&sCp[0]);
    for (int i = threadIdx.x; i < CHUNK * D_ST; i += blockDim.x) {
      const uint32_t pr = (uint32_t)((rbase + (i >> 4)) * NPROJ);
      async_g2l_b32(lB + (uint32_t)i * 4u, (pr + DTRK + (i & 15)) * 4u, proj);
      async_g2l_b32(lC + (uint32_t)i * 4u, (pr + DTRK + D_ST + (i & 15)) * 4u, proj);
    }
  }
  const float alpha = p_alpha[0];
  const float beta  = sigmoidf_dev(p_blogit[0]);
  const float Dp    = D_prm[d];
  float h[D_ST], v[D_ST], Ac[D_ST];
  const size_t cb = ((size_t)(b * NCH + c) * D_IN + d) * 32;
  #pragma unroll
  for (int n = 0; n < D_ST; ++n) {
    h[n] = carry[cb + n];
    v[n] = carry[cb + 16 + n];
    Ac[n] = -__expf(A_log[d * D_ST + n]);
  }
  wait_asynccnt0();
  __syncthreads();
  for (int t = 0; t < CHUNK; ++t) {
    const int r = rbase + t;
    const float dtv = dtbuf[(size_t)r * D_IN + d];
    const float xv  = x[(size_t)r * D_IN + d];
    __builtin_prefetch(&dtbuf[(size_t)(r + 16) * D_IN + d], 0, 0);
    __builtin_prefetch(&x[(size_t)(r + 16) * D_IN + d], 0, 0);
    const float s = alpha * dtv * xv;
    float y = 0.f;
    #pragma unroll
    for (int n = 0; n < D_ST; ++n) {
      const float at = __expf(dtv * Ac[n]);
      v[n] = beta * v[n] + s * sBp[t * D_ST + n];
      h[n] = at * h[n] + v[n];
      y += h[n] * sCp[t * D_ST + n];
    }
    out[(size_t)r * D_IN + d] = y + Dp * xv;
  }
}

// --------------------------------------------------------------- launch --
extern "C" void kernel_launch(void* const* d_in, const int* in_sizes, int n_in,
                              void* d_out, int out_size, void* d_ws, size_t ws_size,
                              hipStream_t stream) {
  (void)in_sizes; (void)n_in; (void)out_size; (void)ws_size;
  const float* x      = (const float*)d_in[0];
  const float* A_log  = (const float*)d_in[1];
  const float* D_prm  = (const float*)d_in[2];
  const float* Wx     = (const float*)d_in[3];
  const float* Wdt    = (const float*)d_in[4];
  const float* b_dt   = (const float*)d_in[5];
  const float* alpha  = (const float*)d_in[6];
  const float* blogit = (const float*)d_in[7];
  float* out = (float*)d_out;

  float* proj  = (float*)d_ws;                               // 8192*64
  float* dtbuf = proj  + (size_t)ROWS * NPROJ;               // 8192*512
  float* summ  = dtbuf + (size_t)ROWS * D_IN;                // 2*64*512*64
  float* carry = summ  + (size_t)BSZ * NCH * D_IN * 64;      // 2*64*512*32
  _Float16* wx16  = (_Float16*)(carry + (size_t)BSZ * NCH * D_IN * 32);
  _Float16* wdt16 = wx16 + (size_t)NPROJ * D_IN;

  prep_kernel<<<(NPROJ * D_IN + D_IN * DTRK + 255) / 256, 256, 0, stream>>>(
      Wx, Wdt, wx16, wdt16);
  gemm_proj<<<ROWS / 128, 256, 0, stream>>>(x, wx16, proj);
  gemm_dt<<<ROWS / 16, 256, 0, stream>>>(proj, wdt16, b_dt, dtbuf);
  scan_phaseA<<<BSZ * NCH, D_IN, 0, stream>>>(x, proj, dtbuf, A_log, alpha,
                                              blogit, summ);
  scan_phaseB<<<(BSZ * D_IN * D_ST) / 256, 256, 0, stream>>>(summ, blogit, carry);
  scan_phaseC<<<BSZ * NCH, D_IN, 0, stream>>>(x, proj, dtbuf, A_log, D_prm,
                                              alpha, blogit, carry, out);
}
